// SelfAttention_87960930222122
// MI455X (gfx1250) — matrix operand order repaired and checked by reading, not compiled, not run on hardware
//
#include <hip/hip_runtime.h>
#include <hip/hip_bf16.h>

// ---- types ----
typedef __bf16 bf16_t;
typedef __bf16 v16bf __attribute__((ext_vector_type(16)));
typedef __bf16 v8bf  __attribute__((ext_vector_type(8)));
typedef __bf16 v4bf  __attribute__((ext_vector_type(4)));
typedef float  v8f   __attribute__((ext_vector_type(8)));

// ---- problem constants (match reference: B=32, S=2048, D=256) ----
#define NB       32
#define NS       2048
#define ND       256
#define TILE_Q   128     // queries per workgroup (8 waves x 16)
#define TI       32      // keys per chunk (one WMMA K-dim for the P*V GEMM)
#define K_STRIDE 280     // LDS row stride (bf16 elems) for [key][d]; 560B -> conflict-free, 16B aligned
#define VT_STRIDE 40     // LDS row stride (bf16 elems) for [d][key]; 80B  -> conflict-free, 16B aligned
#define EPSI     1e-7f

__global__ __launch_bounds__(256)
void selfattn_flash_bf16(const float* __restrict__ x,
                         const int*   __restrict__ mask,
                         float*       __restrict__ out)
{
    // LDS staging of the current 32-key chunk, in both layouts (K = V = x rows)
    __shared__ bf16_t ldsK [TI * K_STRIDE];   // row-major  [key][d]   (A-frags of GEMM1)
    __shared__ bf16_t ldsVT[ND * VT_STRIDE];  // transposed [d][key]   (B-frags of GEMM2)

    const int tid  = threadIdx.x;
    const int wid  = tid >> 5;      // wave id 0..7
    const int lane = tid & 31;
    const int ln   = lane & 15;     // low nibble: N index / M index in frags
    const int hi   = lane >> 4;     // lane half

    const int b  = blockIdx.x >> 4;         // batch
    const int jt = blockIdx.x & 15;         // query tile in batch
    const int q0 = jt * TILE_Q + wid * 16;  // this wave's 16 query rows

    const size_t xb = (size_t)b * NS * ND;

    // ---------- Q fragments, resident in registers ----------
    // GEMM1 is T[key,q] = K(16x32d) * Q^T(32d x 16q); Q is the B operand:
    // lane n = q = ln, b[e] = Q[q, dc*32 + (e<8 ? hi*8+e : 16+hi*8+e-8)]
    v16bf qfrag[8];
    {
        const float* qrow = x + xb + (size_t)(q0 + ln) * ND;
        #pragma unroll
        for (int dc = 0; dc < 8; ++dc) {
            #pragma unroll
            for (int e = 0; e < 16; ++e)
                qfrag[dc][e] = (bf16_t)qrow[dc * 32 + (e < 8 ? hi * 8 + e : 16 + hi * 8 + (e - 8))];
        }
    }

    v16bf ones;
    #pragma unroll
    for (int e = 0; e < 16; ++e) ones[e] = (bf16_t)1.0f;

    // acc[q, d] over 16 d-tiles of 16; lsum[q] (replicated across N columns)
    v8f acc[16];
    #pragma unroll
    for (int dt = 0; dt < 16; ++dt) {
        #pragma unroll
        for (int e = 0; e < 8; ++e) acc[dt][e] = 0.0f;
    }
    v8f lsum;
    #pragma unroll
    for (int e = 0; e < 8; ++e) lsum[e] = 0.0f;

    // staging coordinates: 256 threads cover 32 keys x 256 d (8 float4 each)
    const int skey  = tid & 31;
    const int dpart = tid >> 5;   // 0..7 -> d base = dpart*32

    for (int i0 = 0; i0 < NS; i0 += TI) {
        __syncthreads();   // previous chunk's compute done before overwrite
        {
            const float4* src =
                (const float4*)(x + xb + (size_t)(i0 + skey) * ND + dpart * 32);
            #pragma unroll
            for (int u = 0; u < 8; ++u) {
                float4 v = src[u];
                int d = dpart * 32 + u * 4;
                v4bf pk;
                pk[0] = (bf16_t)v.x; pk[1] = (bf16_t)v.y;
                pk[2] = (bf16_t)v.z; pk[3] = (bf16_t)v.w;
                *(v4bf*)&ldsK[skey * K_STRIDE + d] = pk;          // ds_store_b64
                ldsVT[(d + 0) * VT_STRIDE + skey] = pk[0];        // transposed scatter
                ldsVT[(d + 1) * VT_STRIDE + skey] = pk[1];
                ldsVT[(d + 2) * VT_STRIDE + skey] = pk[2];
                ldsVT[(d + 3) * VT_STRIDE + skey] = pk[3];
            }
            if (i0 + TI < NS)  // hint next chunk toward L2 (global_prefetch_b8)
                __builtin_prefetch(x + xb + (size_t)(i0 + TI + skey) * ND + dpart * 32, 0, 0);
        }
        __syncthreads();

        // ---------- GEMM1: T[key, q] = K_chunk . Q^T  (transposed scores) ----------
        v8f t0, t1;
        #pragma unroll
        for (int e = 0; e < 8; ++e) { t0[e] = 0.0f; t1[e] = 0.0f; }
        #pragma unroll
        for (int dc = 0; dc < 8; ++dc) {
            // A-frag: lane m = key = ln (+16 for second tile);
            // a[0..7]=d@(dc*32+hi*8), a[8..15]=d@(dc*32+16+hi*8)  (both 16B aligned)
            const bf16_t* p0 = &ldsK[ ln        * K_STRIDE + dc * 32 + hi * 8];
            const bf16_t* p1 = &ldsK[(16 + ln)  * K_STRIDE + dc * 32 + hi * 8];
            v8bf a0lo = *(const v8bf*)(p0);
            v8bf a0hi = *(const v8bf*)(p0 + 16);
            v8bf a1lo = *(const v8bf*)(p1);
            v8bf a1hi = *(const v8bf*)(p1 + 16);
            v16bf A0, A1;
            #pragma unroll
            for (int e = 0; e < 8; ++e) {
                A0[e] = a0lo[e]; A0[8 + e] = a0hi[e];
                A1[e] = a1lo[e]; A1[8 + e] = a1hi[e];
            }
            t0 = __builtin_amdgcn_wmma_f32_16x16x32_bf16(false, A0, false, qfrag[dc],
                                                         (short)0, t0, false, false);
            t1 = __builtin_amdgcn_wmma_f32_16x16x32_bf16(false, A1, false, qfrag[dc],
                                                         (short)0, t1, false, false);
        }

        // ---------- exp -> P fragment (A-layout 16q x 32k), lane-local remap ----------
        // C-layout of t0: lane(q=ln,hi) holds keys 8*hi+0..7   == P a[0..7]
        // C-layout of t1: lane(q=ln,hi) holds keys 16+8*hi+0..7 == P a[8..15]
        v16bf P;
        #pragma unroll
        for (int e = 0; e < 8; ++e) {
            P[e]     = (bf16_t)__expf(t0[e]);
            P[8 + e] = (bf16_t)__expf(t1[e]);
        }

        // row-sum via WMMA against ones: every column of D gets sum_k P[q,k]
        lsum = __builtin_amdgcn_wmma_f32_16x16x32_bf16(false, P, false, ones,
                                                       (short)0, lsum, false, false);

        // ---------- GEMM2: acc[q, d] += P . V_chunk ----------
        #pragma unroll
        for (int dt = 0; dt < 16; ++dt) {
            // B-frag: lane n = d = dt*16+ln ; b[0..7] = V[key=hi*8+e, d], b[8..15] = V[key=16+hi*8+e, d]
            const bf16_t* vp = &ldsVT[(dt * 16 + ln) * VT_STRIDE + hi * 8];
            v8bf blo = *(const v8bf*)(vp);
            v8bf bhi = *(const v8bf*)(vp + 16);
            v16bf Bv;
            #pragma unroll
            for (int e = 0; e < 8; ++e) { Bv[e] = blo[e]; Bv[8 + e] = bhi[e]; }
            acc[dt] = __builtin_amdgcn_wmma_f32_16x16x32_bf16(false, P, false, Bv,
                                                              (short)0, acc[dt], false, false);
        }
    }

    // ---------- finalize: out[q,:] = mask[q] ? acc/(lsum+eps) : 0 ----------
    const int* mrow = mask + b * NS + q0;
    float scale[8];
    #pragma unroll
    for (int r = 0; r < 8; ++r) {
        int q = r + 8 * hi;
        scale[r] = mrow[q] ? (1.0f / (lsum[r] + EPSI)) : 0.0f;
    }
    float* orow = out + (size_t)b * NS * ND + (size_t)q0 * ND;
    #pragma unroll
    for (int dt = 0; dt < 16; ++dt) {
        #pragma unroll
        for (int r = 0; r < 8; ++r) {
            int q = r + 8 * hi;
            orow[(size_t)q * ND + dt * 16 + ln] = acc[dt][r] * scale[r];
        }
    }
}

extern "C" void kernel_launch(void* const* d_in, const int* in_sizes, int n_in,
                              void* d_out, int out_size, void* d_ws, size_t ws_size,
                              hipStream_t stream) {
    (void)in_sizes; (void)n_in; (void)out_size; (void)d_ws; (void)ws_size;
    const float* x    = (const float*)d_in[0];
    const int*   mask = (const int*)d_in[1];
    float*       out  = (float*)d_out;
    dim3 grid(NB * (NS / TILE_Q));   // 32 * 16 = 512 workgroups
    dim3 block(256);                 // 8 wave32
    hipLaunchKernelGGL(selfattn_flash_bf16, grid, block, 0, stream, x, mask, out);
}
